// Global_working_space_9835475108028
// MI455X (gfx1250) — compile-verified, matrix operand order checked by reading, not run on hardware
//
#include <hip/hip_runtime.h>
#include <hip/hip_bf16.h>

// Problem constants (from reference): B=2048, NS=S=8, H=512, WH=1, WK=32, RH=4, RK=32
#define BB   2048
#define HH   512

typedef __attribute__((ext_vector_type(16))) __bf16 v16bf;
typedef __attribute__((ext_vector_type(8)))  float  v8f;

__device__ __forceinline__ float sigmoid_f(float x) {
    return 1.0f / (1.0f + __expf(-x));
}

// pack 2 floats -> 2 bf16 with a single v_perm_b32 (take high 16 bits of each)
__device__ __forceinline__ unsigned pack2bf(float lo, float hi) {
    return __builtin_amdgcn_perm(__float_as_uint(hi), __float_as_uint(lo), 0x07060302u);
}
__device__ __forceinline__ uint2 pack4bf(float4 f) {
    return make_uint2(pack2bf(f.x, f.y), pack2bf(f.z, f.w));
}

// async global->LDS copy of 16 bytes per active lane (ASYNCcnt tracked)
__device__ __forceinline__ void async_g2l_b128(void* lds_ptr, const void* gptr) {
    unsigned lds = (unsigned)(unsigned long long)lds_ptr;   // flat->LDS offset
    asm volatile("global_load_async_to_lds_b128 %0, %1, off"
                 :: "v"(lds), "v"(gptr) : "memory");
}
__device__ __forceinline__ void wait_async0() {
    asm volatile("s_wait_asynccnt 0" ::: "memory");
}

// ---------------------------------------------------------------------------
// Generic WMMA GEMM:  out = act( A[M,K] * W[N,K]^T + bias[N] (+ res[M,N]) )
// Block = 256 threads (8 waves, wave32). Block tile 32(M) x 256(N), K step 32.
// Waves arranged 2(M) x 4(N); each wave computes a 16x64 output strip:
// one shared A fragment feeding FOUR v_wmma_f32_16x16x32_bf16 per K step.
// Double-buffered LDS: one __syncthreads per K step; next tile's global
// loads are issued before the current WMMAs so VMEM overlaps matrix ops.
//   out_row = (m / rps) * segStride + segOff + (m % rps)
// act: 0 = none, 1 = relu, 2 = sigmoid
// ---------------------------------------------------------------------------
__global__ __launch_bounds__(256)
void gemm_bf16_wmma(const float* __restrict__ A, const float* __restrict__ W,
                    const float* __restrict__ bias, const float* __restrict__ res,
                    float* __restrict__ out, int M, int N, int K,
                    int rps, int segStride, int segOff, int act)
{
    // pitch 40 halves (80 B): 8-byte aligned rows, conflict-free lane strides
    __shared__ __align__(16) __bf16 As[2][32][40];
    __shared__ __align__(16) __bf16 Ws[2][256][40];

    const int tid  = threadIdx.x;
    const int w    = tid >> 5;          // wave id 0..7
    const int lane = tid & 31;
    const int hl   = lane >> 4;         // half-wave select
    const int l15  = lane & 15;
    const int tileM = blockIdx.y * 32;
    const int tileN = blockIdx.x * 256;
    const int mrow  = ((w >> 2) << 4) + l15;   // A-tile row for this lane
    const int ncolB = ((w & 3) << 6) + l15;    // base W-tile row (4 subtiles, +16 apart)

    // staging coordinates
    const int ar = tid >> 3;            // 0..31  A row
    const int ac = (tid & 7) << 2;      // 0..28  A col (x4 floats)
    const int wr = tid;                 // 0..255 W row (32 floats per row)
    const bool wact = (tileN + wr) < N;

    v8f acc[4] = {};

    float4 a4;
    float4 w4[8];

    auto load_regs = [&](int k0) {
        a4 = *(const float4*)(A + (size_t)(tileM + ar) * K + (k0 + ac));
        if (wact) {
            const float* wp = W + (size_t)(tileN + wr) * K + k0;
#pragma unroll
            for (int i = 0; i < 8; ++i) w4[i] = *(const float4*)(wp + 4 * i);
        }
    };
    auto store_tiles = [&](int buf) {
        *(uint2*)&As[buf][ar][ac] = pack4bf(a4);
        if (wact) {
#pragma unroll
            for (int i = 0; i < 8; ++i)
                *(uint2*)&Ws[buf][wr][4 * i] = pack4bf(w4[i]);
        }
    };

    load_regs(0);
    store_tiles(0);
    __syncthreads();

    int buf = 0;
    for (int k0 = 0; k0 < K; k0 += 32) {
        const bool more = (k0 + 32) < K;
        if (more) load_regs(k0 + 32);   // overlap next tile's VMEM with WMMA

        // fragments per ISA 16-bit 16x32 layout:
        // VGPR vi<4: K in [0,16), vi>=4: K in [16,32); upper half-wave: +8
        v16bf af, bf[4];
#pragma unroll
        for (int vi = 0; vi < 8; ++vi) {
            int kl = ((vi & 4) << 2) + (hl << 3) + ((vi & 3) << 1);
            af[2 * vi]     = As[buf][mrow][kl];
            af[2 * vi + 1] = As[buf][mrow][kl + 1];
#pragma unroll
            for (int t = 0; t < 4; ++t) {
                bf[t][2 * vi]     = Ws[buf][ncolB + 16 * t][kl];
                bf[t][2 * vi + 1] = Ws[buf][ncolB + 16 * t][kl + 1];
            }
        }
#pragma unroll
        for (int t = 0; t < 4; ++t)
            acc[t] = __builtin_amdgcn_wmma_f32_16x16x32_bf16(
                         false, af, false, bf[t], (short)0, acc[t], false, false);

        if (more) {
            store_tiles(buf ^ 1);
            __syncthreads();
            buf ^= 1;
        }
    }

    // ---- epilogue: C/D layout -> lane = N column, VGPR r = row (+8 upper) ----
    const int mbase = tileM + ((w >> 2) << 4) + (hl << 3);
#pragma unroll
    for (int t = 0; t < 4; ++t) {
        const int n = tileN + ncolB + 16 * t;
        if (n < N) {
            const float bv = bias ? bias[n] : 0.0f;
#pragma unroll
            for (int r = 0; r < 8; ++r) {
                int m = mbase + r;
                float val = acc[t][r] + bv;
                if (res) val += res[(size_t)m * N + n];
                if (act == 1)      val = fmaxf(val, 0.0f);
                else if (act == 2) val = sigmoid_f(val);
                size_t orow = (size_t)(m / rps) * segStride + segOff + (m % rps);
                out[orow * (size_t)N + n] = val;
            }
        }
    }
}

// ---------------------------------------------------------------------------
// Write attention: per batch. q:(8,32) k:(16,32) v:(16,512) -> ctx:(8,512)
// V tile (32 KB) staged via GLOBAL_LOAD_ASYNC_TO_LDS_B128, overlapping the
// QK^T + softmax work; fenced with s_wait_asynccnt 0 before first use.
// ---------------------------------------------------------------------------
__global__ __launch_bounds__(256)
void attn_write(const float* __restrict__ qb, const float* __restrict__ kb,
                const float* __restrict__ vb, float* __restrict__ ctx)
{
    const int b = blockIdx.x, tid = threadIdx.x;
    __shared__ float q[8 * 32];
    __shared__ float k[16 * 32];
    __shared__ float sc[8 * 16];
    __shared__ __align__(16) float v[16 * 512];

    // kick off async V staging first: 2048 x 16B chunks, 8 per thread
#pragma unroll
    for (int i = 0; i < 8; ++i) {
        int c16 = tid + (i << 8);                  // 0..2047
        async_g2l_b128(&v[c16 << 2], &vb[(size_t)b * 8192 + (c16 << 2)]);
    }

    q[tid] = qb[(size_t)b * 256 + tid];
#pragma unroll
    for (int i = 0; i < 2; ++i) {
        int idx = tid + (i << 8);
        k[idx] = kb[(size_t)b * 512 + idx];
    }
    __syncthreads();

    if (tid < 128) {
        int si = tid >> 4, ti = tid & 15;
        float a = 0.0f;
        for (int c = 0; c < 32; ++c) a += q[si * 32 + c] * k[ti * 32 + c];
        sc[tid] = a * 0.17677669529663687f;   // 1/sqrt(32)
    }
    __syncthreads();
    if (tid < 8) {
        float mx = -1e30f;
        for (int t = 0; t < 16; ++t) mx = fmaxf(mx, sc[tid * 16 + t]);
        float sum = 0.0f;
        for (int t = 0; t < 16; ++t) {
            float e = __expf(sc[tid * 16 + t] - mx);
            sc[tid * 16 + t] = e; sum += e;
        }
        float inv = 1.0f / sum;
        for (int t = 0; t < 16; ++t) sc[tid * 16 + t] *= inv;
    }
    wait_async0();          // each wave's async V chunks have landed
    __syncthreads();        // -> all waves' chunks visible
#pragma unroll
    for (int i = 0; i < 16; ++i) {
        int idx = tid + (i << 8);          // 0..4095
        int si = idx >> 9, d = idx & 511;
        float a = 0.0f;
        for (int t = 0; t < 16; ++t) a += sc[si * 16 + t] * v[t * 512 + d];
        ctx[(size_t)b * 4096 + idx] = a;
    }
}

// ---------------------------------------------------------------------------
// inputs_mean (relu already applied by GEMM epilogue) + tanh(mem) gate input
// ---------------------------------------------------------------------------
__global__ __launch_bounds__(256)
void mean_kk(const float* __restrict__ ut, const float* __restrict__ mem,
             float* __restrict__ kk)
{
    const int b = blockIdx.x, tid = threadIdx.x;
#pragma unroll
    for (int j = 0; j < 2; ++j) {
        int d = tid + (j << 8);
        float s = 0.0f;
        for (int i = 0; i < 8; ++i) s += ut[((size_t)b * 8 + i) * 512 + d];
        float mean = s * 0.125f;
        for (int i = 0; i < 8; ++i) {
            size_t idx = ((size_t)b * 8 + i) * 512 + d;
            kk[idx] = mean + tanhf(mem[idx]);
        }
    }
}

// ---------------------------------------------------------------------------
// mem_new = sigmoid-gates (pre-applied) : ig*tanh(mem_write) + fg*mem
// ---------------------------------------------------------------------------
__global__ __launch_bounds__(256)
void gate_update(const float* __restrict__ g, const float* __restrict__ mw,
                 const float* __restrict__ mem, float* __restrict__ mn)
{
    size_t idx = (size_t)blockIdx.x * 256 + threadIdx.x;   // exact: B*8*512
    size_t row = idx >> 9;
    int d = (int)(idx & 511);
    float ig = g[row * 1024 + d];
    float fg = g[row * 1024 + 512 + d];
    mn[idx] = ig * tanhf(mw[idx]) + fg * mem[idx];
}

// ---------------------------------------------------------------------------
// Read attention: per batch, 4 heads. q,k:(8,4*32) v:(8,4*512) -> ctx:(8,2048)
// V staged per 1024-column half (32 KB) via async LDS loads; half 1 is only
// issued after the barrier ending half-0 reads (vh is reused).
// ---------------------------------------------------------------------------
__global__ __launch_bounds__(256)
void attn_read(const float* __restrict__ rq, const float* __restrict__ rk,
               const float* __restrict__ rv, float* __restrict__ ctx)
{
    const int b = blockIdx.x, tid = threadIdx.x;
    __shared__ float q[1024];
    __shared__ float k[1024];
    __shared__ float p[256];                    // (h,n,s) row-major
    __shared__ __align__(16) float vh[8192];    // 8 x 1024 column-half of V

    auto issue_half = [&](int chalf) {
#pragma unroll
        for (int i = 0; i < 8; ++i) {
            int c16 = tid + (i << 8);           // 0..2047 16B chunks
            int idx = c16 << 2;                 // float index 0..8188
            int s = idx >> 10, c = idx & 1023;
            async_g2l_b128(&vh[idx],
                           &rv[((size_t)b * 8 + s) * 2048 + chalf * 1024 + c]);
        }
    };

    issue_half(0);   // overlap half-0 staging with QK^T + softmax

#pragma unroll
    for (int i = 0; i < 4; ++i) {
        int idx = tid + (i << 8);
        q[idx] = rq[(size_t)b * 1024 + idx];
        k[idx] = rk[(size_t)b * 1024 + idx];
    }
    __syncthreads();
    {
        int h = tid >> 6, n2 = (tid >> 3) & 7, s = tid & 7;
        float a = 0.0f;
        for (int c = 0; c < 32; ++c)
            a += q[n2 * 128 + h * 32 + c] * k[s * 128 + h * 32 + c];
        p[tid] = a * 0.17677669529663687f;   // 1/sqrt(32)
    }
    __syncthreads();
    if (tid < 32) {                 // 32 softmax rows of length 8
        int base = tid * 8;
        float mx = -1e30f;
        for (int s = 0; s < 8; ++s) mx = fmaxf(mx, p[base + s]);
        float sum = 0.0f;
        for (int s = 0; s < 8; ++s) {
            float e = __expf(p[base + s] - mx);
            p[base + s] = e; sum += e;
        }
        float inv = 1.0f / sum;
        for (int s = 0; s < 8; ++s) p[base + s] *= inv;
    }
    __syncthreads();

    for (int chalf = 0; chalf < 2; ++chalf) {
        wait_async0();              // this half's chunks landed (own wave)
        __syncthreads();            // -> visible across waves
#pragma unroll
        for (int i = 0; i < 32; ++i) {
            int idx = tid + (i << 8);
            int n2 = idx >> 10, c = idx & 1023;
            int h = chalf * 2 + (c >> 9);
            float a = 0.0f;
            for (int s = 0; s < 8; ++s)
                a += p[(h * 8 + n2) * 8 + s] * vh[s * 1024 + c];
            ctx[((size_t)b * 8 + n2) * 2048 + chalf * 1024 + c] = a;
        }
        __syncthreads();            // all reads of vh done before overwrite
        if (chalf == 0) issue_half(1);
    }
}

// ---------------------------------------------------------------------------
extern "C" void kernel_launch(void* const* d_in, const int* in_sizes, int n_in,
                              void* d_out, int out_size, void* d_ws, size_t ws_size,
                              hipStream_t stream)
{
    (void)in_sizes; (void)n_in; (void)out_size; (void)ws_size;

    const float* h_s  = (const float*)d_in[0];
    const float* mem  = (const float*)d_in[1];
    const float* wq_w = (const float*)d_in[2];
    const float* wq_b = (const float*)d_in[3];
    const float* wk_w = (const float*)d_in[4];
    const float* wk_b = (const float*)d_in[5];
    const float* wv_w = (const float*)d_in[6];
    const float* wv_b = (const float*)d_in[7];
    const float* wo_w = (const float*)d_in[8];
    const float* wo_b = (const float*)d_in[9];
    const float* ut_w = (const float*)d_in[10];
    const float* ut_b = (const float*)d_in[11];
    const float* ug_w = (const float*)d_in[12];
    const float* ug_b = (const float*)d_in[13];
    const float* rq_w = (const float*)d_in[14];
    const float* rq_b = (const float*)d_in[15];
    const float* rk_w = (const float*)d_in[16];
    const float* rk_b = (const float*)d_in[17];
    const float* rv_w = (const float*)d_in[18];
    const float* rv_b = (const float*)d_in[19];
    const float* ro_w = (const float*)d_in[20];
    const float* ro_b = (const float*)d_in[21];

    // Workspace layout (floats). ctx2 aliases the phase-1 region, whose
    // buffers are all dead by the time the read-attention kernel writes it.
    float* ws    = (float*)d_ws;
    float* kbuf  = ws;                                  // B*16*32
    float* vbuf  = kbuf  + (size_t)BB * 16 * 32;        // B*16*512
    float* qbuf  = vbuf  + (size_t)BB * 16 * 512;       // B*8*32
    float* ctx1  = qbuf  + (size_t)BB * 8 * 32;         // B*8*512
    float* utbuf = ctx1  + (size_t)BB * 8 * 512;        // B*8*512
    float* kkbuf = utbuf + (size_t)BB * 8 * 512;        // B*8*512
    float* gbuf  = kkbuf + (size_t)BB * 8 * 512;        // B*8*1024
    float* mwbuf = gbuf  + (size_t)BB * 8 * 1024;       // B*8*512
    float* rvbuf = mwbuf + (size_t)BB * 8 * 512;        // B*8*2048
    float* rqbuf = rvbuf + (size_t)BB * 8 * 2048;       // B*8*128
    float* rkbuf = rqbuf + (size_t)BB * 8 * 128;        // B*8*128
    float* ctx2  = ws;                                  // B*8*2048 (alias)

    float* hout   = (float*)d_out;
    float* memnew = hout + (size_t)BB * 8 * HH;         // second output

    const int M = BB * 8;   // 16384 rows in every flattened GEMM

    auto gemm = [&](const float* A, const float* W, const float* bias,
                    const float* res, float* out, int N, int K,
                    int segStride, int segOff, int act) {
        dim3 grid((N + 255) / 256, M / 32);
        gemm_bf16_wmma<<<grid, 256, 0, stream>>>(A, W, bias, res, out,
                                                 M, N, K, 8, segStride, segOff, act);
    };

    // ---- write attention projections (K/V interleave h_s and mem rows) ----
    gemm(h_s, wk_w, wk_b, nullptr, kbuf, 32,  512, 16, 0, 0);  // k rows 0..7
    gemm(mem, wk_w, wk_b, nullptr, kbuf, 32,  512, 16, 8, 0);  // k rows 8..15
    gemm(h_s, wv_w, wv_b, nullptr, vbuf, 512, 512, 16, 0, 0);  // v rows 0..7
    gemm(mem, wv_w, wv_b, nullptr, vbuf, 512, 512, 16, 8, 0);  // v rows 8..15
    gemm(mem, wq_w, wq_b, nullptr, qbuf, 32,  512,  8, 0, 0);
    gemm(h_s, ut_w, ut_b, nullptr, utbuf,512, 512,  8, 0, 1);  // fused relu

    attn_write<<<BB, 256, 0, stream>>>(qbuf, kbuf, vbuf, ctx1);
    mean_kk<<<BB, 256, 0, stream>>>(utbuf, mem, kkbuf);

    gemm(kkbuf, ug_w, ug_b, nullptr, gbuf,  1024, 512, 8, 0, 2);  // fused sigmoid
    gemm(ctx1,  wo_w, wo_b, nullptr, mwbuf, 512,  512, 8, 0, 0);

    gate_update<<<(M * 512) / 256, 256, 0, stream>>>(gbuf, mwbuf, mem, memnew);

    // ---- read attention projections (rk/rv consume mem_new from d_out) ----
    gemm(h_s,    rq_w, rq_b, nullptr, rqbuf, 128,  512, 8, 0, 0);
    gemm(memnew, rk_w, rk_b, nullptr, rkbuf, 128,  512, 8, 0, 0);
    gemm(memnew, rv_w, rv_b, nullptr, rvbuf, 2048, 512, 8, 0, 0);

    attn_read<<<BB, 256, 0, stream>>>(rqbuf, rkbuf, rvbuf, ctx2);

    // h_out = h_s + ctx2 * ro^T + ro_b   (residual fused in epilogue)
    gemm(ctx2, ro_w, ro_b, h_s, hout, 512, 2048, 8, 0, 0);
}
